// WTRPNet_83330955477840
// MI455X (gfx1250) — matrix operand order repaired and checked by reading, not compiled, not run on hardware
//
#include <hip/hip_runtime.h>
#include <stdint.h>

// ---------------------------------------------------------------------------
// MI455X (gfx1250), wave32. Matrix path: v_wmma_f32_16x16x32_bf16.
// All WMMA operand streams are 16B-aligned vector loads:
//   - A activations: bf16 row-major with K padded to 32 and 16B-aligned rows
//   - conv inputs:   bf16 NHWC with K-order (di*4+dj)*Cin + c  -> contiguous
//   - B weights:     pre-packed into the exact wave32 fragment layout (32B/lane)
// Workspace: ~215 MB, L2-resident on MI455X (192 MB L2).
// ---------------------------------------------------------------------------

typedef __attribute__((ext_vector_type(16))) __bf16 v16bf;
typedef __attribute__((ext_vector_type(8)))  __bf16 v8bf;
typedef __attribute__((ext_vector_type(8)))  float  v8f;

static constexpr int   Bb      = 512;
static constexpr int   Nn      = 100;
static constexpr int   Mm      = 16;
static constexpr int   Kk      = 4;
static constexpr float SIGMA2  = 0.001f;
static constexpr float R_MIN_C = 1.0f;
static constexpr float P_MAX_C = 10.0f;
static constexpr float P_BIAS_C= 0.0f;
static constexpr int   NUM_ITER= 10;
static constexpr float LR_W_C  = 1.0e-4f;
static constexpr float LR_T_C  = 1.0e-3f;
static constexpr float LR_R_C  = 1.0e-3f;
static constexpr float MOM_C   = 0.9f;
static constexpr float BN_SC   = 0.99999500003749973f; // 1/sqrt(1+1e-5)

__device__ __forceinline__ v16bf cat8(v8bf lo, v8bf hi) {
    return __builtin_shufflevector(lo, hi, 0,1,2,3,4,5,6,7,8,9,10,11,12,13,14,15);
}

// ---------------------------------------------------------------------------
// x (B x 4000) bf16:  x[b, n*40 + c*2 + p]
// ---------------------------------------------------------------------------
__global__ void build_x_kernel(const float* __restrict__ Hr, const float* __restrict__ Hi,
                               const float* __restrict__ Gr, const float* __restrict__ Gi,
                               __bf16* __restrict__ x)
{
    int idx = blockIdx.x * blockDim.x + threadIdx.x;
    if (idx >= Bb * 4000) return;
    int b = idx / 4000;
    int r = idx - b * 4000;
    int n = r / 40;
    int t = r - n * 40;
    int c = t >> 1, p = t & 1;
    float v;
    if (c < Kk) v = p ? Hi[(b * Nn + n) * Kk + c] : Hr[(b * Nn + n) * Kk + c];
    else        v = p ? Gi[(b * Nn + n) * Mm + (c - Kk)] : Gr[(b * Nn + n) * Mm + (c - Kk)];
    x[idx] = (__bf16)v;
}

// ---------------------------------------------------------------------------
// tnet conv input, NHWC bf16 (B, 16, 100, 8):
//   casc[b,k,m,n] = conj(H[b,n,k])*G[b,n,m]; channel ch = 2k + (0:re,1:im)
// ---------------------------------------------------------------------------
__global__ void build_xt_kernel(const float* __restrict__ Hr, const float* __restrict__ Hi,
                                const float* __restrict__ Gr, const float* __restrict__ Gi,
                                __bf16* __restrict__ xt)
{
    int idx = blockIdx.x * blockDim.x + threadIdx.x;
    if (idx >= Bb * 16 * 100 * 8) return;
    int ch = idx & 7;
    int t2 = idx >> 3;
    int n  = t2 % 100;
    int t3 = t2 / 100;
    int m  = t3 & 15;
    int b  = t3 >> 4;
    int k = ch >> 1, p = ch & 1;
    float hr = Hr[(b * Nn + n) * Kk + k], hi = Hi[(b * Nn + n) * Kk + k];
    float gr = Gr[(b * Nn + n) * Mm + m], gi = Gi[(b * Nn + n) * Mm + m];
    float v = p ? (hr * gi - hi * gr) : (hr * gr + hi * gi);
    xt[idx] = (__bf16)v;
}

// ---------------------------------------------------------------------------
// Pack f32 weights (Kreal x Nreal row-major) into the wave32 bf16 B-fragment
// layout, zero-padded to KT*32 x NT*16:
//   tile (nt,kt): lane L -> col n = nt*16 + L%16, h = L/16; element i -> K =
//   kt*32 + h*8 + (i&7) + (i>>3)*16.  Contiguous 32B per lane.
// ---------------------------------------------------------------------------
__global__ void pack_b_kernel(const float* __restrict__ src, int Kreal, int Nreal,
                              int KT, int NT, __bf16* __restrict__ dst)
{
    int idx = blockIdx.x * blockDim.x + threadIdx.x;
    if (idx >= KT * NT * 32) return;
    int lane = idx & 31;
    int tile = idx >> 5;               // tile = nt*KT + kt
    int kt = tile % KT;
    int nt = tile / KT;
    int n  = nt * 16 + (lane & 15);
    int h  = lane >> 4;
    __bf16* out = dst + ((size_t)tile * 32 + lane) * 16;
#pragma unroll
    for (int i = 0; i < 16; ++i) {
        int k = kt * 32 + h * 8 + (i & 7) + (i >> 3) * 16;
        float v = (k < Kreal && n < Nreal) ? src[(size_t)k * Nreal + n] : 0.f;
        out[i] = (__bf16)v;
    }
}

// ---------------------------------------------------------------------------
// Pack conv weights src (64, Cin*16) with K-order kk = (di*4+dj)*Cin + c.
// src element for (o, kk): src[o*(Cin*16) + c*16 + pos], pos = di*4+dj.
// ---------------------------------------------------------------------------
__global__ void pack_conv_kernel(const float* __restrict__ src, int Cin,
                                 int KT, __bf16* __restrict__ dst)
{
    const int NT = 4;
    int idx = blockIdx.x * blockDim.x + threadIdx.x;
    if (idx >= KT * NT * 32) return;
    int lane = idx & 31;
    int tile = idx >> 5;
    int kt = tile % KT;
    int nt = tile / KT;
    int o  = nt * 16 + (lane & 15);
    int h  = lane >> 4;
    __bf16* out = dst + ((size_t)tile * 32 + lane) * 16;
#pragma unroll
    for (int i = 0; i < 16; ++i) {
        int kk  = kt * 32 + h * 8 + (i & 7) + (i >> 3) * 16;
        int c   = kk % Cin;
        int pos = kk / Cin;
        out[i] = (__bf16)src[(size_t)o * (Cin * 16) + c * 16 + pos];
    }
}

// ---------------------------------------------------------------------------
// WMMA GEMM: C = A(Mrows x Kp bf16 row-major, Kp = KT*32, 16B-aligned rows)
//              * B(packed).  MREP row-tiles per wave share each B fragment.
// Epilogues: 0 +bias | 1 +bias,BN+ReLU | 2 +bias,ReLU+R_MIN | 3 sigmoid*P_MAX
// outH (if set) is zero-padded to ldH = NT*16 columns.
// ---------------------------------------------------------------------------
__global__ void gemm_wmma_kernel(const __bf16* __restrict__ A, int lda,
                                 const __bf16* __restrict__ Bp,
                                 int Mrows, int Ncols, int KT, int NT,
                                 const float* __restrict__ bias,
                                 const float* __restrict__ gamma,
                                 const float* __restrict__ beta,
                                 int epi,
                                 float* __restrict__ outF, int ldF,
                                 __bf16* __restrict__ outH, int ldH)
{
    constexpr int MREP = 2;
    int wave = (blockIdx.x * blockDim.x + threadIdx.x) >> 5;
    int lane = threadIdx.x & 31;
    int MT = (Mrows + 15) >> 4;
    int MG = (MT + MREP - 1) / MREP;
    if (wave >= MG * NT) return;
    int mg = wave / NT, nt = wave - mg * NT;
    int r = lane & 15, h = lane >> 4;

    size_t rowOff[MREP];
#pragma unroll
    for (int u = 0; u < MREP; ++u) {
        int row = (mg * MREP + u) * 16 + r;
        row = row < Mrows ? row : Mrows - 1;
        rowOff[u] = (size_t)row * lda;
    }

    v8f acc[MREP];
#pragma unroll
    for (int u = 0; u < MREP; ++u) acc[u] = (v8f){0.f,0.f,0.f,0.f,0.f,0.f,0.f,0.f};

    const __bf16* bptr = Bp + ((size_t)nt * KT * 32 + lane) * 16;

    for (int kt = 0; kt < KT; ++kt) {
        v16bf bmat = *(const v16bf*)bptr;
        __builtin_prefetch(bptr + 512, 0, 3);           // near-scope prefetch
        int kbase = kt * 32 + h * 8;                    // 16B-aligned offsets
#pragma unroll
        for (int u = 0; u < MREP; ++u) {
            const __bf16* ap = A + rowOff[u] + kbase;
            v8bf lo = *(const v8bf*)ap;                 // K = h*8 .. h*8+7
            v8bf hi = *(const v8bf*)(ap + 16);          // K = 16+h*8 ..
            acc[u] = __builtin_amdgcn_wmma_f32_16x16x32_bf16(
                false, cat8(lo, hi), false, bmat, (short)0, acc[u], false, false);
        }
        bptr += 512;
    }

    int n = nt * 16 + r;
#pragma unroll
    for (int u = 0; u < MREP; ++u) {
#pragma unroll
        for (int vv = 0; vv < 8; ++vv) {
            int m = (mg * MREP + u) * 16 + vv + 8 * h;
            if (m >= Mrows) continue;
            float xv = 0.f;
            if (n < Ncols) {
                xv = acc[u][vv] + (bias ? bias[n] : 0.f);
                if (epi == 1)      { xv = gamma[n] * (xv * BN_SC) + beta[n]; xv = fmaxf(xv, 0.f); }
                else if (epi == 2) { xv = fmaxf(xv, 0.f) + R_MIN_C; }
                else if (epi == 3) { xv = P_MAX_C / (1.f + __expf(-(xv + P_BIAS_C))); }
                if (outF) outF[(size_t)m * ldF + n] = xv;
            }
            if (outH) outH[(size_t)m * ldH + n] = (__bf16)xv;  // n < NT*16 == ldH
        }
    }
}

// ---------------------------------------------------------------------------
// Implicit-GEMM 4x4 VALID conv, 64 out channels, bias+BN+ReLU fused.
// Input NHWC bf16, K-order kk = (di*4+dj)*Cin + c -> every 8-element A run is
// 8 consecutive channels at one spatial position: one aligned 16B load.
// Output NHWC (out_nchw=0) or NCHW (out_nchw=1, == flatten order for dense).
// ---------------------------------------------------------------------------
__global__ void conv_wmma_kernel(const __bf16* __restrict__ in, int CinLog2,
                                 int Hin, int Win, int Ho, int Wo,
                                 const __bf16* __restrict__ Bp, int KT,
                                 const float* __restrict__ bias,
                                 const float* __restrict__ gamma,
                                 const float* __restrict__ beta,
                                 __bf16* __restrict__ out, int out_nchw)
{
    constexpr int MREP = 4;
    const int NT = 4;
    int wave = (blockIdx.x * blockDim.x + threadIdx.x) >> 5;
    int lane = threadIdx.x & 31;
    int rowsPerB = Ho * Wo;
    int Mrows = Bb * rowsPerB;
    int MT = (Mrows + 15) >> 4;
    int MG = (MT + MREP - 1) / MREP;
    if (wave >= MG * NT) return;
    int mg = wave / NT, nt = wave - mg * NT;
    int r = lane & 15, h = lane >> 4;
    int cmask = (1 << CinLog2) - 1;

    // per-rep base address of spatial position (oi, oj) in NHWC
    size_t pbase[MREP];
#pragma unroll
    for (int u = 0; u < MREP; ++u) {
        int row = (mg * MREP + u) * 16 + r;
        row = row < Mrows ? row : Mrows - 1;
        int b  = row / rowsPerB;
        int rem = row - b * rowsPerB;
        int oi = rem / Wo, oj = rem - oi * Wo;
        pbase[u] = ((size_t)(b * Hin + oi) * Win + oj) << CinLog2;
    }

    v8f acc[MREP];
#pragma unroll
    for (int u = 0; u < MREP; ++u) acc[u] = (v8f){0.f,0.f,0.f,0.f,0.f,0.f,0.f,0.f};

    const __bf16* bptr = Bp + ((size_t)nt * KT * 32 + lane) * 16;
    const int WinC = Win << CinLog2;

    for (int kt = 0; kt < KT; ++kt) {
        v16bf bmat = *(const v16bf*)bptr;
        __builtin_prefetch(bptr + 512, 0, 3);           // near-scope prefetch
#pragma unroll
        for (int u = 0; u < MREP; ++u) {
            v8bf half[2];
#pragma unroll
            for (int g = 0; g < 2; ++g) {
                int kb  = kt * 32 + g * 16 + h * 8;
                int c   = kb & cmask;
                int pos = kb >> CinLog2;
                int di  = pos >> 2, dj = pos & 3;
                const __bf16* p = in + pbase[u] + (size_t)di * WinC + ((size_t)dj << CinLog2) + c;
                half[g] = *(const v8bf*)p;              // aligned 16B
            }
            acc[u] = __builtin_amdgcn_wmma_f32_16x16x32_bf16(
                false, cat8(half[0], half[1]), false, bmat, (short)0, acc[u], false, false);
        }
        bptr += 512;
    }

    int o = nt * 16 + r;
#pragma unroll
    for (int u = 0; u < MREP; ++u) {
#pragma unroll
        for (int vv = 0; vv < 8; ++vv) {
            int m = (mg * MREP + u) * 16 + vv + 8 * h;
            if (m >= Mrows) continue;
            int b2 = m / rowsPerB;
            int rem = m - b2 * rowsPerB;
            int i2 = rem / Wo, j2 = rem - i2 * Wo;
            float xv = acc[u][vv] + bias[o];
            xv = gamma[o] * (xv * BN_SC) + beta[o];
            xv = fmaxf(xv, 0.f);
            size_t dst = out_nchw
                ? (((size_t)b2 * 64 + o) * Ho + i2) * Wo + j2
                : (((size_t)b2 * Ho + i2) * Wo + j2) * 64 + o;
            out[dst] = (__bf16)xv;
        }
    }
}

// ---------------------------------------------------------------------------
// Projection: 10 momentum-GD steps on V(vt,vr,w), hand-derived Wirtinger
// gradient. One 128-thread workgroup per batch element, state in LDS (~30KB).
// ---------------------------------------------------------------------------
__global__ void project_kernel(const float* __restrict__ Hre, const float* __restrict__ Him,
                               const float* __restrict__ Gre, const float* __restrict__ Gim,
                               float* __restrict__ out_w, float* __restrict__ out_vt,
                               float* __restrict__ out_vr,
                               const float* __restrict__ rmin, const float* __restrict__ pmax)
{
    int b = blockIdx.x;
    int t = threadIdx.x; // 128 threads

    __shared__ float Hr[Nn * Kk], Hi[Nn * Kk], Gr[Nn * Mm], Gi[Nn * Mm];
    __shared__ float vt[Nn], vrs[Nn], w[129];
    __shared__ float momt[Nn], momr[Nn], momw[129];
    __shared__ float akr[Nn * Kk], aki[Nn * Kk];
    __shared__ float hefr[64], hefi[64];
    __shared__ float er[16], ei[16], ssv[16], gsv[16];
    __shared__ float Fr[64], Fi[64];
    __shared__ float dkv[4], invIA[4], invI[4];
    __shared__ float gw[129];
    __shared__ float sc[4]; // 0:nrm2  1:c  2:Vc  3:p_max
    __shared__ float rm[4];
    __shared__ float Ttr[Nn], Tti[Nn], Trr[Nn], Tri[Nn];

    for (int i = t; i < Nn * Kk; i += blockDim.x) { Hr[i] = Hre[(size_t)b * Nn * Kk + i]; Hi[i] = Him[(size_t)b * Nn * Kk + i]; }
    for (int i = t; i < Nn * Mm; i += blockDim.x) { Gr[i] = Gre[(size_t)b * Nn * Mm + i]; Gi[i] = Gim[(size_t)b * Nn * Mm + i]; }
    for (int i = t; i < Nn; i += blockDim.x) { vt[i] = out_vt[b * Nn + i]; vrs[i] = out_vr[b * Nn + i]; momt[i] = 0.f; momr[i] = 0.f; }
    for (int i = t; i < 129; i += blockDim.x) { w[i] = out_w[b * 129 + i]; momw[i] = 0.f; }
    if (t < 4)  rm[t] = rmin[b * 4 + t];
    if (t == 0) sc[3] = pmax[b];
    __syncthreads();

    const float inv_ln2 = 1.4426950408889634f;

    for (int it = 0; it < NUM_ITER; ++it) {
        for (int i = t; i < Nn; i += blockDim.x) {
            float s, c;
            __sincosf(vt[i], &s, &c);  Ttr[i] = 0.70710678f * c; Tti[i] = 0.70710678f * s;
            __sincosf(vrs[i], &s, &c); Trr[i] = 0.70710678f * c; Tri[i] = 0.70710678f * s;
        }
        __syncthreads();
        for (int i = t; i < Nn * Kk; i += blockDim.x) {
            int n = i / Kk, k = i - n * Kk;
            float tr = (k < 2) ? Ttr[n] : Trr[n];
            float ti = (k < 2) ? Tti[n] : Tri[n];
            akr[i] = Hr[i] * tr + Hi[i] * ti;
            aki[i] = Hr[i] * ti - Hi[i] * tr;
        }
        if (t == 0) {
            float s = 0.f;
            for (int i = 0; i < 128; ++i) s += w[i] * w[i];
            sc[0] = s;
            sc[1] = sqrtf(sc[3]) / (sqrtf(s) + 1e-12f);
        }
        __syncthreads();
        if (t < 64) {
            int k = t >> 4, m = t & 15;
            float sr = 0.f, si = 0.f;
            for (int n = 0; n < Nn; ++n) {
                float ar = akr[n * Kk + k], ai = aki[n * Kk + k];
                float gr = Gr[n * Mm + m], gi = Gi[n * Mm + m];
                sr += ar * gr - ai * gi;
                si += ar * gi + ai * gr;
            }
            hefr[t] = sr; hefi[t] = si;
        }
        __syncthreads();
        float cc = sc[1];
        if (t < 16) {
            int k = t >> 2, j = t & 3;
            float sr = 0.f, si = 0.f;
            for (int m = 0; m < 16; ++m) {
                float wr = w[m * 4 + j], wi = w[64 + m * 4 + j];
                float hr = hefr[k * 16 + m], hi = hefi[k * 16 + m];
                sr += hr * wr - hi * wi;
                si += hr * wi + hi * wr;
            }
            er[t] = cc * sr; ei[t] = cc * si;
            ssv[t] = er[t] * er[t] + ei[t] * ei[t];
        }
        __syncthreads();
        if (t < 4) {
            float A = ssv[t * 4 + t];
            float I = SIGMA2 - A;
            for (int j = 0; j < 4; ++j) I += ssv[t * 4 + j];
            float R = log2f(1.f + A / I);
            dkv[t]   = fmaxf(rm[t] - R + 1e-5f, 0.f);
            invIA[t] = 1.f / (I + A);
            invI[t]  = 1.f / I;
        }
        __syncthreads();
        if (t < 16) {
            int k = t >> 2, j = t & 3;
            gsv[t] = -2.f * dkv[k] * inv_ln2 *
                     ((j == k) ? invIA[k] : (invIA[k] - invI[k]));
        }
        __syncthreads();
        if (t == 0) {
            float vcs = 0.f;
            for (int i = 0; i < 16; ++i) vcs += gsv[i] * 2.f * ssv[i];
            sc[2] = vcs / cc;
        }
        __syncthreads();
        float Vc = sc[2], nrm2 = sc[0];
        if (t < 64) {
            int m = t >> 2, j = t & 3;
            float gr = 0.f, gi = 0.f;
            for (int k = 0; k < 4; ++k) {
                float g  = gsv[k * 4 + j];
                float e_r = er[k * 4 + j], e_i = ei[k * 4 + j];
                float h_r = hefr[k * 16 + m], h_i = hefi[k * 16 + m];
                gr += g * (e_r * h_r + e_i * h_i);
                gi += g * (e_i * h_r - e_r * h_i);
            }
            float cof = Vc * cc / nrm2;
            gw[t]      = 2.f * cc * gr - cof * w[t];
            gw[64 + t] = 2.f * cc * gi - cof * w[64 + t];
            int k2 = t >> 4, m2 = t & 15;
            float fr = 0.f, fi = 0.f;
            for (int j2 = 0; j2 < 4; ++j2) {
                float g  = gsv[k2 * 4 + j2];
                float e_r = er[k2 * 4 + j2], e_i = ei[k2 * 4 + j2];
                float wr = w[m2 * 4 + j2], wi = w[64 + m2 * 4 + j2];
                fr += g * (e_r * wr + e_i * wi);
                fi += g * (e_r * wi - e_i * wr);
            }
            Fr[t] = cc * fr; Fi[t] = cc * fi;
        }
        if (t == 64) gw[128] = 0.f;
        __syncthreads();
        if (t < Nn) {
            float gvt = 0.f, gvr = 0.f;
            for (int k = 0; k < 4; ++k) {
                float Qr = 0.f, Qi = 0.f;
                for (int m = 0; m < 16; ++m) {
                    float fr = Fr[k * 16 + m], fi = Fi[k * 16 + m];
                    float gr = Gr[t * Mm + m], gi = Gi[t * Mm + m];
                    Qr += fr * gr - fi * gi;
                    Qi += fr * gi + fi * gr;
                }
                float hr = Hr[t * Kk + k], hi = Hi[t * Kk + k];
                float zr = hr * Qr + hi * Qi;
                float zi = hr * Qi - hi * Qr;
                if (k < 2) gvt += -2.f * (Ttr[t] * zi + Tti[t] * zr);
                else       gvr += -2.f * (Trr[t] * zi + Tri[t] * zr);
            }
            float st = LR_T_C * gvt + MOM_C * momt[t];
            float sr = LR_R_C * gvr + MOM_C * momr[t];
            vt[t]  -= st; momt[t] = st;
            vrs[t] -= sr; momr[t] = sr;
        }
        if (t < 129) {
            float sw = LR_W_C * gw[t] + MOM_C * momw[t];
            w[t] -= sw; momw[t] = sw;
        }
        __syncthreads();
    }

    for (int i = t; i < Nn; i += blockDim.x) { out_vt[b * Nn + i] = vt[i]; out_vr[b * Nn + i] = vrs[i]; }
    for (int i = t; i < 129; i += blockDim.x) out_w[b * 129 + i] = w[i];
}

// ---------------------------------------------------------------------------
// Host orchestration
// ---------------------------------------------------------------------------
extern "C" void kernel_launch(void* const* d_in, const int* in_sizes, int n_in,
                              void* d_out, int out_size, void* d_ws, size_t ws_size,
                              hipStream_t stream)
{
    (void)in_sizes; (void)n_in; (void)out_size; (void)ws_size;

    const float* H_re = (const float*)d_in[0];
    const float* H_im = (const float*)d_in[1];
    const float* G_re = (const float*)d_in[2];
    const float* G_im = (const float*)d_in[3];
    const float* c1w = (const float*)d_in[4];
    const float* c1b = (const float*)d_in[5];
    const float* tg1 = (const float*)d_in[6];
    const float* tbe1= (const float*)d_in[7];
    const float* c2w = (const float*)d_in[8];
    const float* c2b = (const float*)d_in[9];
    const float* tg2 = (const float*)d_in[10];
    const float* tbe2= (const float*)d_in[11];
    const float* tw1 = (const float*)d_in[12];
    const float* tb1 = (const float*)d_in[13];
    const float* tg3 = (const float*)d_in[14];
    const float* tbe3= (const float*)d_in[15];
    const float* tw2 = (const float*)d_in[16];
    const float* tb2 = (const float*)d_in[17];
    const float* tw3 = (const float*)d_in[18];
    const float* tb3 = (const float*)d_in[19];
    const float* mp[3][10];
    for (int net = 0; net < 3; ++net)
        for (int i = 0; i < 10; ++i)
            mp[net][i] = (const float*)d_in[20 + net * 10 + i];

    float* out  = (float*)d_out;
    float* o_w  = out;                 // 512 x 129
    float* o_vt = o_w  + Bb * 129;     // 512 x 100
    float* o_vr = o_vt + Bb * Nn;      // 512 x 100
    float* o_rm = o_vr + Bb * Nn;      // 512 x 4
    float* o_pm = o_rm + Bb * 4;       // 512 x 1

    uint8_t* ws = (uint8_t*)d_ws;
    size_t off = 0;
    auto alloc = [&](size_t bytes) -> void* {
        void* p = ws + off;
        off = (off + bytes + 255) & ~(size_t)255;
        return p;
    };

    // activations (bf16, padded K / aligned rows)
    __bf16* x   = (__bf16*)alloc((size_t)Bb * 4000 * 2);           // K=4000
    __bf16* xt  = (__bf16*)alloc((size_t)Bb * 16 * 100 * 8 * 2);   // NHWC
    __bf16* c1o = (__bf16*)alloc((size_t)Bb * 13 * 97 * 64 * 2);   // NHWC
    __bf16* c2o = (__bf16*)alloc((size_t)Bb * 64 * 10 * 94 * 2);   // NCHW = flat
    __bf16* ht  = (__bf16*)alloc((size_t)Bb * 416 * 2);            // 400 -> 416
    __bf16* h1  = (__bf16*)alloc((size_t)Bb * 128 * 2);            // 100 -> 128
    __bf16* h2  = (__bf16*)alloc((size_t)Bb * 128 * 2);

    auto packBytes = [](int KT, int NT) { return (size_t)KT * NT * 1024; };
    __bf16* c1p  = (__bf16*)alloc(packBytes(4, 4));      // K=128
    __bf16* c2p  = (__bf16*)alloc(packBytes(32, 4));     // K=1024
    __bf16* tw1p = (__bf16*)alloc(packBytes(1880, 26));  // 60160 x 400->416
    __bf16* tw2p = (__bf16*)alloc(packBytes(13, 7));     // 416 x 100
    __bf16* tw3p = (__bf16*)alloc(packBytes(13, 7));
    __bf16* mw1p[3], *mw2p[3], *mw3p[3];
    const int outDim3[3] = {129, 4, 1};
    const int NT3[3]     = {9, 1, 1};
    for (int net = 0; net < 3; ++net) {
        mw1p[net] = (__bf16*)alloc(packBytes(125, 8));         // 4000 x 100->128
        mw2p[net] = (__bf16*)alloc(packBytes(4, 8));           // 128 x 100->128
        mw3p[net] = (__bf16*)alloc(packBytes(4, NT3[net]));    // 128 x out
    }

    auto pack = [&](const float* src, int Kreal, int Nreal, int KT, int NT, __bf16* dst) {
        int threads = KT * NT * 32;
        pack_b_kernel<<<(threads + 255) / 256, 256, 0, stream>>>(src, Kreal, Nreal, KT, NT, dst);
    };
    auto gemm = [&](const __bf16* A, int lda, const __bf16* Bp,
                    int Mr, int Nc, int KT, int NT,
                    const float* bias, const float* g, const float* be, int epi,
                    float* oF, int ldF, __bf16* oH, int ldH) {
        int MG = (((Mr + 15) / 16) + 1) / 2;   // MREP = 2
        int tiles = MG * NT;
        gemm_wmma_kernel<<<(tiles + 7) / 8, 256, 0, stream>>>(
            A, lda, Bp, Mr, Nc, KT, NT, bias, g, be, epi, oF, ldF, oH, ldH);
    };

    // 1. inputs
    build_x_kernel <<<(Bb * 4000 + 255) / 256, 256, 0, stream>>>(H_re, H_im, G_re, G_im, x);
    build_xt_kernel<<<(Bb * 12800 + 255) / 256, 256, 0, stream>>>(H_re, H_im, G_re, G_im, xt);

    // 2. weight packing
    {
        int th1 = 4 * 4 * 32;
        pack_conv_kernel<<<(th1 + 255) / 256, 256, 0, stream>>>(c1w, 8, 4, c1p);
        int th2 = 32 * 4 * 32;
        pack_conv_kernel<<<(th2 + 255) / 256, 256, 0, stream>>>(c2w, 64, 32, c2p);
    }
    pack(tw1, 60160, 400, 1880, 26, tw1p);
    pack(tw2, 400, 100, 13, 7, tw2p);
    pack(tw3, 400, 100, 13, 7, tw3p);
    for (int net = 0; net < 3; ++net) {
        pack(mp[net][0], 4000, 100, 125, 8, mw1p[net]);
        pack(mp[net][4], 100, 100, 4, 8, mw2p[net]);
        pack(mp[net][8], 100, outDim3[net], 4, NT3[net], mw3p[net]);
    }

    // 3. convs (implicit GEMM, bias+BN+ReLU fused); MREP=4
    {
        int Mr = Bb * 13 * 97;
        int tiles = ((((Mr + 15) / 16) + 3) / 4) * 4;
        conv_wmma_kernel<<<(tiles + 7) / 8, 256, 0, stream>>>(
            xt, 3, 16, 100, 13, 97, c1p, 4, c1b, tg1, tbe1, c1o, 0);
    }
    {
        int Mr = Bb * 10 * 94;
        int tiles = ((((Mr + 15) / 16) + 3) / 4) * 4;
        conv_wmma_kernel<<<(tiles + 7) / 8, 256, 0, stream>>>(
            c1o, 6, 13, 97, 10, 94, c2p, 32, c2b, tg2, tbe2, c2o, 1);
    }

    // 4. tnet dense trunk (c2o NCHW contiguous == flatten order)
    gemm(c2o, 60160, tw1p, Bb, 400, 1880, 26, tb1, tg3, tbe3, 1, nullptr, 0, ht, 416);
    gemm(ht,  416,   tw2p, Bb, 100, 13, 7, tb2, nullptr, nullptr, 0, o_vt, 100, nullptr, 0);
    gemm(ht,  416,   tw3p, Bb, 100, 13, 7, tb3, nullptr, nullptr, 0, o_vr, 100, nullptr, 0);

    // 5. wnet / rnet / pnet trunks
    float* head_out[3] = {o_w, o_rm, o_pm};
    int    head_ld[3]  = {129, 4, 1};
    int    head_epi[3] = {0, 2, 3};
    for (int net = 0; net < 3; ++net) {
        gemm(x,  4000, mw1p[net], Bb, 100, 125, 8,
             mp[net][1], mp[net][2], mp[net][3], 1, nullptr, 0, h1, 128);
        gemm(h1, 128,  mw2p[net], Bb, 100, 4, 8,
             mp[net][5], mp[net][6], mp[net][7], 1, nullptr, 0, h2, 128);
        gemm(h2, 128,  mw3p[net], Bb, outDim3[net], 4, NT3[net],
             mp[net][9], nullptr, nullptr, head_epi[net],
             head_out[net], head_ld[net], nullptr, 0);
    }

    // 6. projection (refines w / var_t / var_r in place in d_out)
    project_kernel<<<Bb, 128, 0, stream>>>(H_re, H_im, G_re, G_im,
                                           o_w, o_vt, o_vr, o_rm, o_pm);
}
